// MultiHeadAttention_66451734004245
// MI455X (gfx1250) — matrix operand order repaired and checked by reading, not compiled, not run on hardware
//
#include <hip/hip_runtime.h>

typedef __attribute__((ext_vector_type(16))) __bf16 v16bf;
typedef __attribute__((ext_vector_type(8)))  float  v8f;
typedef __attribute__((ext_vector_type(4)))  unsigned int u32x4;
typedef __attribute__((ext_vector_type(4)))  float  f32x4;

#define CD    512   // channels
#define WINT  64    // tokens per window
#define HDIM  64    // head dim

// LDS layout (bytes). Total 272 KB of the 320 KB WGP pool.
#define LDS_X   0        // 64*512 bf16 = 65536; reused as probs[8][64*64] bf16 in stage 3
#define LDS_Q   65536    // row-major  (token x 512) bf16
#define LDS_K   131072   // row-major  (token x 512) bf16
#define LDS_V   196608   // COLUMN-major (channel x 64 tokens) bf16
#define LDS_BB  262144   // 64*64 f32 = 16384
#define LDS_TOTAL 278528

#define WELEM   (512 * 512)                 // elements per weight matrix
#define WBYTES  (3ull * WELEM * 2ull)       // bf16 scratch bytes needed

static __device__ inline v8f vzero8() {
  v8f z;
#pragma unroll
  for (int i = 0; i < 8; ++i) z[i] = 0.0f;
  return z;
}

static __device__ inline v8f wmma_bf16(v16bf a, v16bf b, v8f c) {
  // D = A(16x32 bf16) x B(32x16 bf16) + C(16x16 f32)
  return __builtin_amdgcn_wmma_f32_16x16x32_bf16(false, a, false, b,
                                                 (short)0, c, false, false);
}

// A-fragment (16x32 bf16) from a row-major bf16 LDS matrix, leading dim ldm.
// Layout: lane l (m = l&15), elems 0..15 -> k = kbase+lg*16+e (contiguous).
static __device__ inline v16bf lds_load_A(const char* smem, int off, int ldm,
                                          int m, int kbase, int lg) {
  const int k0 = kbase + lg * 16;
  union { u32x4 u[2]; v16bf v; } cv;
  cv.u[0] = *(const u32x4*)(smem + off + (m * ldm + k0) * 2);
  cv.u[1] = *(const u32x4*)(smem + off + (m * ldm + k0 + 8) * 2);
  return cv.v;
}

// B-fragment (32x16) where B[k][n] = M[n][k]: lane n reads 16 contiguous bf16
// of row n of M (k = kbase + lg*16 + e).
static __device__ inline v16bf lds_load_Brow(const char* smem, int off, int ldm,
                                             int n, int kbase, int lg) {
  const int k0 = kbase + lg * 16;
  union { u32x4 u[2]; v16bf v; } cv;
  const u32x4* p = (const u32x4*)(smem + off + (n * ldm + k0) * 2);
  cv.u[0] = p[0];
  cv.u[1] = p[1];
  return cv.v;
}

// B-fragment for X @ W^T from pre-converted bf16 W: two global_load_b128.
static __device__ inline v16bf g_load_Bw16(const __bf16* __restrict__ Wm,
                                           int n, int kbase, int lg) {
  const u32x4* p = (const u32x4*)(Wm + n * CD + kbase + lg * 16);
  union { u32x4 u[2]; v16bf v; } cv;
  cv.u[0] = p[0];
  cv.u[1] = p[1];
  return cv.v;
}

// Fallback: B-fragment from fp32 W with inline conversion.
static __device__ inline v16bf g_load_Bw32(const float* __restrict__ Wm,
                                           int n, int kbase, int lg) {
  const float* p = Wm + n * CD + kbase + lg * 16;
  v16bf r;
#pragma unroll
  for (int e = 0; e < 16; ++e) r[e] = (__bf16)p[e];
  return r;
}

// ---- pre-pass: Wq|Wk|Wv fp32 -> bf16 into d_ws (run once per launch) -----
__global__ __launch_bounds__(256)
void mha_convert_weights(const float* __restrict__ Wq,
                         const float* __restrict__ Wk,
                         const float* __restrict__ Wv,
                         __bf16* __restrict__ dst) {
  const int i = (blockIdx.x * 256 + threadIdx.x) * 4;   // grid covers 3*WELEM
  const float* src = (i < WELEM)     ? Wq + i
                   : (i < 2 * WELEM) ? Wk + (i - WELEM)
                                     : Wv + (i - 2 * WELEM);
  const f32x4 f = *(const f32x4*)src;
  union { __bf16 h[4]; unsigned long long u; } t;
  t.h[0] = (__bf16)f[0]; t.h[1] = (__bf16)f[1];
  t.h[2] = (__bf16)f[2]; t.h[3] = (__bf16)f[3];
  *(unsigned long long*)(dst + i) = t.u;
}

template <bool BF16W>
__global__ __launch_bounds__(256, 1)
void MultiHeadAttention_66451734004245_kernel(
    const float* __restrict__ x,
    const float* __restrict__ Wq, const float* __restrict__ bq,
    const float* __restrict__ Wk, const float* __restrict__ bk,
    const float* __restrict__ Wv, const float* __restrict__ bv,
    const float* __restrict__ Bbias,
    const __bf16* __restrict__ Wqb, const __bf16* __restrict__ Wkb,
    const __bf16* __restrict__ Wvb,
    float* __restrict__ out) {
  extern __shared__ char smem[];
  const int tid  = threadIdx.x;
  const int wave = tid >> 5;        // 8 wave32 per block
  const int lane = tid & 31;
  const int l15  = lane & 15;
  const int lg   = lane >> 4;

  // (b, window) tile is contiguous: 64 tokens x 512 channels, row-major.
  const long base = (long)blockIdx.x * (WINT * CD);
  const float* __restrict__ xw   = x + base;
  float* __restrict__       outw = out + base;

  // ---- Stage 1: X fp32 -> bf16 LDS; Bbias -> fp32 LDS -------------------
  for (int i = tid * 4; i < WINT * CD; i += 256 * 4) {
    const f32x4 f = *(const f32x4*)(xw + i);
    union { __bf16 h[4]; unsigned long long u; } t;
    t.h[0] = (__bf16)f[0]; t.h[1] = (__bf16)f[1];
    t.h[2] = (__bf16)f[2]; t.h[3] = (__bf16)f[3];
    *(unsigned long long*)(smem + LDS_X + (long)i * 2) = t.u;
  }
  for (int i = tid; i < WINT * WINT; i += 256)
    ((float*)(smem + LDS_BB))[i] = Bbias[i];
  __syncthreads();

  // ---- Stage 2: Q/K/V = X @ W^T + b  (wave owns 64 output channels) -----
  const int ncol0 = wave * 64;
#pragma unroll 1
  for (int mat = 0; mat < 3; ++mat) {
    const float*  Wm  = (mat == 0) ? Wq  : (mat == 1) ? Wk  : Wv;
    const __bf16* Wmb = (mat == 0) ? Wqb : (mat == 1) ? Wkb : Wvb;
    const float*  bm  = (mat == 0) ? bq  : (mat == 1) ? bk  : bv;

    v8f acc[4][4];
#pragma unroll
    for (int mt = 0; mt < 4; ++mt)
#pragma unroll
      for (int nt = 0; nt < 4; ++nt) acc[mt][nt] = vzero8();

#pragma unroll 1
    for (int kc = 0; kc < 16; ++kc) {
      const int kbase = kc * 32;
      v16bf a[4];
#pragma unroll
      for (int mt = 0; mt < 4; ++mt)
        a[mt] = lds_load_A(smem, LDS_X, CD, mt * 16 + l15, kbase, lg);
#pragma unroll
      for (int nt = 0; nt < 4; ++nt) {
        const int n_ = ncol0 + nt * 16 + l15;
        v16bf bw;
        if constexpr (BF16W) bw = g_load_Bw16(Wmb, n_, kbase, lg);
        else                 bw = g_load_Bw32(Wm, n_, kbase, lg);
#pragma unroll
        for (int mt = 0; mt < 4; ++mt)
          acc[mt][nt] = wmma_bf16(a[mt], bw, acc[mt][nt]);
      }
      // hint next weight chunk toward cache (global_prefetch_b8)
      if constexpr (BF16W)
        __builtin_prefetch(Wmb + (ncol0 + l15) * CD + ((kc + 1) & 15) * 32, 0, 1);
      else
        __builtin_prefetch(Wm + (ncol0 + l15) * CD + ((kc + 1) & 15) * 32, 0, 1);
    }

    if (mat == 2) {
      // V: column-major (channel x token). Tokens are consecutive in r ->
      // pack 8 bf16 and do one ds_store_b128 per tile.
#pragma unroll
      for (int nt = 0; nt < 4; ++nt) {
        const int n = ncol0 + nt * 16 + l15;           // channel
        const float bias = bm[n];
#pragma unroll
        for (int mt = 0; mt < 4; ++mt) {
          union { __bf16 h[8]; u32x4 u; } t;
#pragma unroll
          for (int r = 0; r < 8; ++r)
            t.h[r] = (__bf16)(acc[mt][nt][r] + bias);
          *(u32x4*)(smem + LDS_V + (n * WINT + mt * 16 + 8 * lg) * 2) = t.u;
        }
      }
    } else {
      // Q, K: row-major (token x 512); D-layout rows are strided -> b16 stores.
      const int ldsOut = (mat == 0) ? LDS_Q : LDS_K;
#pragma unroll
      for (int nt = 0; nt < 4; ++nt) {
        const int n = ncol0 + nt * 16 + l15;
        const float bias = bm[n];
#pragma unroll
        for (int mt = 0; mt < 4; ++mt)
#pragma unroll
          for (int r = 0; r < 8; ++r) {
            const int m = mt * 16 + r + 8 * lg;        // token
            *(__bf16*)(smem + ldsOut + (m * CD + n) * 2) =
                (__bf16)(acc[mt][nt][r] + bias);
          }
      }
    }
  }
  __syncthreads();

  // ---- Stage 3: per-head attention (wave == head), fully transposed -----
  const int h  = wave;
  const int hc = h * HDIM;

  // S^T = K @ Q^T : s[at][bt]; rows kk (key token) in r, cols q (query) in lane.
  v8f s[4][4];
#pragma unroll
  for (int at = 0; at < 4; ++at)
#pragma unroll
    for (int bt = 0; bt < 4; ++bt) s[at][bt] = vzero8();

#pragma unroll
  for (int kc = 0; kc < 2; ++kc) {          // K-dim = 64 head channels
    const int kbase = hc + kc * 32;
    v16bf a[4];
#pragma unroll
    for (int at = 0; at < 4; ++at)
      a[at] = lds_load_A(smem, LDS_K, CD, at * 16 + l15, kbase, lg);
#pragma unroll
    for (int bt = 0; bt < 4; ++bt) {
      const v16bf bq_ = lds_load_Brow(smem, LDS_Q, CD, bt * 16 + l15, kbase, lg);
#pragma unroll
      for (int at = 0; at < 4; ++at)
        s[at][bt] = wmma_bf16(a[at], bq_, s[at][bt]);
    }
  }

  // scale + relative-position bias: S^T[kk][q] += bb[q*64+kk]; kk consecutive
  // in r -> two contiguous 16B LDS loads per tile.
  const float scale = 0.125f;  // 1/sqrt(64)
  const float* bb = (const float*)(smem + LDS_BB);
#pragma unroll
  for (int at = 0; at < 4; ++at)
#pragma unroll
    for (int bt = 0; bt < 4; ++bt) {
      const int q   = bt * 16 + l15;
      const int kk0 = at * 16 + 8 * lg;
      const f32x4 b0 = *(const f32x4*)(bb + q * WINT + kk0);
      const f32x4 b1 = *(const f32x4*)(bb + q * WINT + kk0 + 4);
#pragma unroll
      for (int r = 0; r < 4; ++r) s[at][bt][r] = s[at][bt][r] * scale + b0[r];
#pragma unroll
      for (int r = 0; r < 4; ++r) s[at][bt][r + 4] = s[at][bt][r + 4] * scale + b1[r];
    }

  // softmax over kk (rows of S^T) per column q: 32 values in-lane, the other
  // 32 in the paired lane (lane ^ 16) -> one shfl_xor per reduction.
#pragma unroll
  for (int bt = 0; bt < 4; ++bt) {
    float mx = -3.4e38f;
#pragma unroll
    for (int at = 0; at < 4; ++at)
#pragma unroll
      for (int r = 0; r < 8; ++r) mx = fmaxf(mx, s[at][bt][r]);
    mx = fmaxf(mx, __shfl_xor(mx, 16, 32));
    float sum = 0.0f;
#pragma unroll
    for (int at = 0; at < 4; ++at)
#pragma unroll
      for (int r = 0; r < 8; ++r) {
        const float e = __expf(s[at][bt][r] - mx);
        s[at][bt][r] = e;
        sum += e;
      }
    sum += __shfl_xor(sum, 16, 32);
    const float inv = 1.0f / sum;
#pragma unroll
    for (int at = 0; at < 4; ++at)
#pragma unroll
      for (int r = 0; r < 8; ++r) s[at][bt][r] *= inv;
  }

  // P row-major bf16 (P[q][kk], q = token): kk consecutive in r -> packed
  // ds_store_b128 into the retired X region (8 KB per head).
  const int pbase = LDS_X + h * (WINT * WINT * 2);
#pragma unroll
  for (int bt = 0; bt < 4; ++bt) {
    const int q = bt * 16 + l15;
#pragma unroll
    for (int at = 0; at < 4; ++at) {
      const int kk0 = at * 16 + 8 * lg;
      union { __bf16 hh[8]; u32x4 u; } t;
#pragma unroll
      for (int r = 0; r < 8; ++r) t.hh[r] = (__bf16)s[at][bt][r];
      *(u32x4*)(smem + pbase + (q * WINT + kk0) * 2) = t.u;
    }
  }
  __syncthreads();

  // O^T = V^T @ P^T : A rows = head channels from column-major V (contiguous
  // tokens -> ds_load_b128), B rows = rows of row-major P.
  v8f o[4][4];
#pragma unroll
  for (int at = 0; at < 4; ++at)
#pragma unroll
    for (int bt = 0; bt < 4; ++bt) o[at][bt] = vzero8();

#pragma unroll
  for (int kc = 0; kc < 2; ++kc) {          // K-dim = 64 key tokens
    const int kbase = kc * 32;
    v16bf a[4];
#pragma unroll
    for (int at = 0; at < 4; ++at)          // row = head channel
      a[at] = lds_load_A(smem, LDS_V, WINT, hc + at * 16 + l15, kbase, lg);
#pragma unroll
    for (int bt = 0; bt < 4; ++bt) {        // col = output token
      const v16bf bp_ = lds_load_Brow(smem, pbase, WINT, bt * 16 + l15, kbase, lg);
#pragma unroll
      for (int at = 0; at < 4; ++at)
        o[at][bt] = wmma_bf16(a[at], bp_, o[at][bt]);
    }
  }

  // store: O^T rows are channels (consecutive in r) -> per tile two packed
  // global_store_b128 at out[token*512 + hc + channel].
#pragma unroll
  for (int bt = 0; bt < 4; ++bt) {
    const int m = bt * 16 + l15;            // token
#pragma unroll
    for (int at = 0; at < 4; ++at) {
      float* p = outw + m * CD + hc + at * 16 + 8 * lg;
      union { v8f v; f32x4 f4[2]; } t;
      t.v = o[at][bt];
      *(f32x4*)p       = t.f4[0];
      *((f32x4*)p + 1) = t.f4[1];
    }
  }
}

extern "C" void kernel_launch(void* const* d_in, const int* in_sizes, int n_in,
                              void* d_out, int out_size, void* d_ws, size_t ws_size,
                              hipStream_t stream) {
  (void)in_sizes; (void)n_in; (void)out_size;
  const float* x  = (const float*)d_in[0];
  const float* Wq = (const float*)d_in[1];
  const float* bq = (const float*)d_in[2];
  const float* Wk = (const float*)d_in[3];
  const float* bk = (const float*)d_in[4];
  const float* Wv = (const float*)d_in[5];
  const float* bv = (const float*)d_in[6];
  const float* Bb = (const float*)d_in[7];
  float* out = (float*)d_out;

  const dim3 grid(8 * 64);   // B * NW windows
  const dim3 block(256);     // 8 wave32

  if (ws_size >= WBYTES) {
    // One-time fp32 -> bf16 weight conversion into scratch (L2-resident).
    __bf16* wsb = (__bf16*)d_ws;
    mha_convert_weights<<<3 * WELEM / 1024, 256, 0, stream>>>(Wq, Wk, Wv, wsb);

    hipFuncSetAttribute((const void*)&MultiHeadAttention_66451734004245_kernel<true>,
                        hipFuncAttributeMaxDynamicSharedMemorySize, LDS_TOTAL);
    MultiHeadAttention_66451734004245_kernel<true>
        <<<grid, block, LDS_TOTAL, stream>>>(
            x, Wq, bq, Wk, bk, Wv, bv, Bb,
            wsb, wsb + WELEM, wsb + 2 * WELEM, out);
  } else {
    hipFuncSetAttribute((const void*)&MultiHeadAttention_66451734004245_kernel<false>,
                        hipFuncAttributeMaxDynamicSharedMemorySize, LDS_TOTAL);
    MultiHeadAttention_66451734004245_kernel<false>
        <<<grid, block, LDS_TOTAL, stream>>>(
            x, Wq, bq, Wk, bk, Wv, bv, Bb,
            nullptr, nullptr, nullptr, out);
  }
}
